// MultiHeadAttention_8126078124475
// MI455X (gfx1250) — matrix operand order repaired and checked by reading, not compiled, not run on hardware
//
#include <hip/hip_runtime.h>
#include <hip/hip_bf16.h>

#define DIM    1024
#define NHEAD  16
#define HD     64
#define BATCH  2
#define SEQ    2048
#define MTOT   (BATCH * SEQ)   // 4096
#define NQKV   (3 * DIM)       // 3072

typedef __attribute__((ext_vector_type(16))) __bf16       v16bf;
typedef __attribute__((ext_vector_type(8)))  float        v8f;
typedef __attribute__((ext_vector_type(4)))  unsigned int u32x4;
typedef __attribute__((ext_vector_type(8)))  unsigned int u32x8;

union Frag { v16bf v; u32x4 u[2]; };

__device__ __forceinline__ unsigned short f2bf(float f) {
  unsigned int u = __float_as_uint(f);
  u += 0x7FFFu + ((u >> 16) & 1u);   // round-to-nearest-even
  return (unsigned short)(u >> 16);
}

__device__ __forceinline__ v8f vzero8() {
  v8f z;
#pragma unroll
  for (int i = 0; i < 8; ++i) z[i] = 0.0f;
  return z;
}

// flat shared pointer -> 32-bit LDS byte offset (addrspacecast keeps offset in low 32)
__device__ __forceinline__ unsigned int to_lds(const void* p) {
  return (unsigned int)(size_t)p;
}

// ---- CDNA5 async global->LDS copy (ASYNCcnt-tracked, no VGPR round trip) ----
__device__ __forceinline__ void async_ld_b128(unsigned int lds_byte, const void* g) {
  asm volatile("global_load_async_to_lds_b128 %0, %1, off"
               :: "v"(lds_byte), "v"(g) : "memory");
}
__device__ __forceinline__ void wait_async0() {
  asm volatile("s_wait_asynccnt 0x0" ::: "memory");
}

// ---- CDNA5 Tensor Data Mover: 2D bf16 tile (tile_w x tile_h, element stride)
// D# per CDNA5 ISA: group0 = {count, lds_addr, global_addr,
// type=2}; group1 = {data_size=2B, tensor dims, tile dims, dim0 stride}.
__device__ __forceinline__ void tdm_load_2d(unsigned int lds_byte, const void* gptr,
                                            unsigned int tile_w, unsigned int tile_h,
                                            unsigned int stride_elems) {
  const unsigned long long ga = (unsigned long long)gptr;
  u32x4 g0;
  g0[0] = 1u;                                               // count=1, user mode
  g0[1] = lds_byte;                                         // lds_addr
  g0[2] = (unsigned int)ga;                                 // global_addr[31:0]
  g0[3] = (unsigned int)((ga >> 32) & 0x01FFFFFFu) | (2u << 30);  // [56:32] | type=2
  u32x8 g1;
  g1[0] = 0x00010000u;                                      // data_size=1 (2 bytes)
  g1[1] = (tile_w & 0xFFFFu) << 16;                         // tensor_dim0 lo16
  g1[2] = (tile_w >> 16) | ((tile_h & 0xFFFFu) << 16);      // dim0 hi | tensor_dim1 lo16
  g1[3] = (tile_h >> 16) | ((tile_w & 0xFFFFu) << 16);      // dim1 hi | tile_dim0
  g1[4] = tile_h & 0xFFFFu;                                 // tile_dim1 (tile_dim2=0)
  g1[5] = stride_elems;                                     // tensor_dim0_stride lo32
  g1[6] = 0u;                                               // stride hi | dim1_stride
  g1[7] = 0u;
  asm volatile("tensor_load_to_lds %0, %1" :: "s"(g0), "s"(g1) : "memory");
}

// A-fragment (16x32 bf16): lane L holds row (L&15); halves K = g*8+{0..7} in
// v0..3 and K = g*8+16+{0..7} in v4..7, g = L>>4. rowp = this lane's row base.
__device__ __forceinline__ void load_a(Frag& f, const unsigned short* rowp, int g) {
  f.u[0] = *(const u32x4*)(rowp + g * 8);
  f.u[1] = *(const u32x4*)(rowp + g * 8 + 16);
}
// B-fragment (32x16 bf16): lane L holds column (L&15); halves K = g*8+{0..7}
// in v0..3 and K = g*8+16+{0..7} in v4..7. colp = this lane's column base (K-contiguous in LDS/memory).
__device__ __forceinline__ void load_b(Frag& f, const unsigned short* colp, int g) {
  f.u[0] = *(const u32x4*)(colp + g * 8);
  f.u[1] = *(const u32x4*)(colp + g * 8 + 16);
}

__device__ __forceinline__ v8f wmma_bf16(const Frag& a, const Frag& b, v8f c) {
  return __builtin_amdgcn_wmma_f32_16x16x32_bf16(false, a.v, false, b.v,
                                                 (short)0, c, false, false);
}

// ---------------------------------------------------------------------------
// Tiled GEMM: C[M,N] = A[M,K](bf16) * Bt[N,K](bf16, i.e. B col-major) + bias
// Double-buffered: async copies of tile t+1 overlap WMMA on tile t.
// mode 0: write fp32 to outF[M,N].  mode 1: scatter bf16 into q/k/v [B,H,N,HD].
// ---------------------------------------------------------------------------
#define TM 128
#define TN 64
#define TK 32

__device__ __forceinline__ void stage_tiles(const unsigned short* A,
                                            const unsigned short* Bt,
                                            unsigned short* dA, unsigned short* dB,
                                            int bm, int bn, int K, int kk, int tid) {
  // A tile 128x32: 512 chunks of 8 halves, exactly 2 per thread (no exec churn)
#pragma unroll
  for (int it = 0; it < 2; ++it) {
    const int c  = tid + it * 256;
    const int r  = c >> 2;
    const int c8 = (c & 3) * 8;
    async_ld_b128(to_lds(dA + r * TK + c8),
                  A + (size_t)(bm + r) * K + kk + c8);
  }
  // Bt tile 64x32: 256 chunks, 1 per thread
  const int r  = tid >> 2;
  const int c8 = (tid & 3) * 8;
  async_ld_b128(to_lds(dB + r * TK + c8),
                Bt + (size_t)(bn + r) * K + kk + c8);
}

__global__ __launch_bounds__(256)
void gemm_bf16_kernel(const unsigned short* __restrict__ A,
                      const unsigned short* __restrict__ Bt,
                      const float* __restrict__ bias,
                      float* __restrict__ outF,
                      unsigned short* __restrict__ outQKV,
                      int M, int N, int K, int mode) {
  __shared__ __align__(16) unsigned short ldsA[2][TM * TK];
  __shared__ __align__(16) unsigned short ldsB[2][TN * TK];

  const int tid  = threadIdx.x;
  const int wave = tid >> 5;
  const int lane = tid & 31;
  const int cl   = lane & 15;
  const int g    = lane >> 4;

  const int ntiles = N / TN;
  const int bm = (blockIdx.x / ntiles) * TM;
  const int bn = (blockIdx.x % ntiles) * TN;

  v8f acc[4];
#pragma unroll
  for (int j = 0; j < 4; ++j) acc[j] = vzero8();

  int ib = 0;
  stage_tiles(A, Bt, ldsA[0], ldsB[0], bm, bn, K, 0, tid);

  for (int kk = 0; kk < K; kk += TK) {
    wait_async0();          // tile ib fully in LDS (per-wave), then publish
    __syncthreads();
    if (kk + TK < K)        // kick off DMA for tile ib^1; overlaps the math below
      stage_tiles(A, Bt, ldsA[ib ^ 1], ldsB[ib ^ 1], bm, bn, K, kk + TK, tid);
    if (kk + 2 * TK < K) {
      __builtin_prefetch(A + (size_t)(bm + (tid >> 1)) * K + kk + 2 * TK, 0, 1);
      __builtin_prefetch(Bt + (size_t)(bn + (tid & 63)) * K + kk + 2 * TK, 0, 1);
    }

    Frag a;
    load_a(a, &ldsA[ib][(wave * 16 + cl) * TK], g);
#pragma unroll
    for (int j = 0; j < 4; ++j) {
      Frag b;
      load_b(b, &ldsB[ib][(j * 16 + cl) * TK], g);
      acc[j] = wmma_bf16(a, b, acc[j]);
    }
    __syncthreads();        // all waves done reading ib before it is restaged
    ib ^= 1;
  }

  // epilogue: C/D layout -> vgpr i, lanes 0-15 row i, lanes 16-31 row i+8
#pragma unroll
  for (int j = 0; j < 4; ++j) {
    const int col = bn + j * 16 + cl;
    const float bval = bias[col];
#pragma unroll
    for (int i = 0; i < 8; ++i) {
      const int row = bm + wave * 16 + i + 8 * g;
      const float val = acc[j][i] + bval;
      if (mode == 0) {
        outF[(size_t)row * N + col] = val;
      } else {
        const int which = col >> 10;   // 0=q 1=k 2=v
        const int d     = col & 1023;
        const int h     = d >> 6;
        const int hd    = d & 63;
        const int b     = row >> 11;
        const int n     = row & 2047;
        outQKV[(size_t)which * ((size_t)BATCH * NHEAD * SEQ * HD) +
               (((size_t)(b * NHEAD + h) * SEQ + n) * HD + hd)] = f2bf(val);
      }
    }
  }
}

// ---------------------------------------------------------------------------
// Flash attention: block = (b,h, 128 q rows), 8 waves x 16 q rows each.
// Double-buffered K/V: K via Tensor Data Mover, V staged transposed; next
// tile's DMA overlaps softmax + WMMA of the current tile.
// ---------------------------------------------------------------------------
__device__ __forceinline__ void stage_v_tr(const unsigned short* vh,
                                           unsigned short* dVT, int j0, int tid) {
  const int r  = tid >> 3;
  const int c8 = (tid & 7) * 8;
  union { u32x4 v4; unsigned short hx[8]; } tmp;
  tmp.v4 = *(const u32x4*)(vh + (size_t)(j0 + r) * HD + c8);
#pragma unroll
  for (int e = 0; e < 8; ++e) dVT[(c8 + e) * 32 + r] = tmp.hx[e];
}

__global__ __launch_bounds__(256)
void attn_kernel(const unsigned short* __restrict__ q,
                 const unsigned short* __restrict__ k,
                 const unsigned short* __restrict__ v,
                 unsigned short* __restrict__ y) {
  __shared__ __align__(16) unsigned short ldsK[2][32 * HD];   // [kv][d]
  __shared__ __align__(16) unsigned short ldsVT[2][HD * 32];  // [d][kv]
  __shared__ __align__(16) unsigned short ldsP[8][16 * 32];

  const int tid  = threadIdx.x;
  const int wave = tid >> 5;
  const int lane = tid & 31;
  const int cl   = lane & 15;
  const int g    = lane >> 4;

  const int bh    = blockIdx.x >> 4;
  const int qblk  = blockIdx.x & 15;
  const int b     = bh >> 4;
  const int h     = bh & 15;
  const int qrow0 = qblk * 128 + wave * 16;

  const unsigned short* qh = q + (size_t)bh * SEQ * HD;
  const unsigned short* kh = k + (size_t)bh * SEQ * HD;
  const unsigned short* vh = v + (size_t)bh * SEQ * HD;

  Frag aq0, aq1;  // Q fragment 16x64 = two 16x32 A-fragments
  {
    const unsigned short* rp = qh + (size_t)(qrow0 + cl) * HD;
    load_a(aq0, rp, g);
    load_a(aq1, rp + 32, g);
  }

  v8f o[4];
#pragma unroll
  for (int d = 0; d < 4; ++d) o[d] = vzero8();
  float mrow[8], lrow[8];
#pragma unroll
  for (int i = 0; i < 8; ++i) { mrow[i] = -1e30f; lrow[i] = 0.0f; }

  int ib = 0;
  if (wave == 0) tdm_load_2d(to_lds(ldsK[0]), kh, HD, 32, HD);
  stage_v_tr(vh, ldsVT[0], 0, tid);

  for (int j0 = 0; j0 < SEQ; j0 += 32) {
    if (wave == 0) __builtin_amdgcn_s_wait_tensorcnt(0);
    __syncthreads();        // K (TDM) and V (ds stores) of tile ib published
    if (j0 + 32 < SEQ) {    // DMA next tile while computing this one
      if (wave == 0)
        tdm_load_2d(to_lds(ldsK[ib ^ 1]), kh + (size_t)(j0 + 32) * HD, HD, 32, HD);
      stage_v_tr(vh, ldsVT[ib ^ 1], j0 + 32, tid);
    }

    // S(16x32) = Q(16x64) . K_tile(32x64)^T, two 16x16 tiles, 2 WMMAs each
    v8f s[2];
#pragma unroll
    for (int jj = 0; jj < 2; ++jj) {
      Frag bk0, bk1;
      const unsigned short* cp = &ldsK[ib][(jj * 16 + cl) * HD];
      load_b(bk0, cp, g);
      load_b(bk1, cp + 32, g);
      s[jj] = wmma_bf16(aq0, bk0, vzero8());
      s[jj] = wmma_bf16(aq1, bk1, s[jj]);
    }

    // online softmax over these 32 keys; row = i + 8*g within the wave tile
#pragma unroll
    for (int i = 0; i < 8; ++i) {
      const float s0 = s[0][i] * 0.125f;   // 1/sqrt(64)
      const float s1 = s[1][i] * 0.125f;
      float mx = fmaxf(s0, s1);
#pragma unroll
      for (int off = 1; off < 16; off <<= 1)
        mx = fmaxf(mx, __shfl_xor(mx, off, 32));   // xor<16 stays in 16-lane group
      const float mnew  = fmaxf(mrow[i], mx);
      const float alpha = __expf(mrow[i] - mnew);
      const float p0 = __expf(s0 - mnew);
      const float p1 = __expf(s1 - mnew);
      float ps = p0 + p1;
#pragma unroll
      for (int off = 1; off < 16; off <<= 1)
        ps += __shfl_xor(ps, off, 32);
      lrow[i] = lrow[i] * alpha + ps;
      mrow[i] = mnew;
#pragma unroll
      for (int d = 0; d < 4; ++d) o[d][i] *= alpha;
      const int pr = (i + 8 * g) * 32;
      ldsP[wave][pr + cl]      = f2bf(p0);
      ldsP[wave][pr + 16 + cl] = f2bf(p1);
    }

    // O(16x64) += P(16x32) . V_tile(32x64)
    Frag ap;
    load_a(ap, &ldsP[wave][cl * 32], g);
#pragma unroll
    for (int d = 0; d < 4; ++d) {
      Frag bv;
      load_b(bv, &ldsVT[ib][(d * 16 + cl) * 32], g);
      o[d] = wmma_bf16(ap, bv, o[d]);
    }
    __syncthreads();        // reads of tile ib done before it is restaged
    ib ^= 1;
  }

  // normalize + store bf16 y[B,N,DIM]
#pragma unroll
  for (int d = 0; d < 4; ++d) {
#pragma unroll
    for (int i = 0; i < 8; ++i) {
      const int n = qrow0 + i + 8 * g;
      const float val = o[d][i] / lrow[i];
      y[((size_t)(b * SEQ + n)) * DIM + h * HD + d * 16 + cl] = f2bf(val);
    }
  }
}

// ---------------------------------------------------------------------------
__global__ void cvt_kernel(const float* __restrict__ in,
                           unsigned short* __restrict__ out, int n) {
  const int i = blockIdx.x * blockDim.x + threadIdx.x;
  if (i < n) out[i] = f2bf(in[i]);
}

// wt[n*K + k] = bf16(w[k*N + n])  (transpose so B fragments are K-contiguous)
__global__ void transpose_cvt_kernel(const float* __restrict__ w,
                                     unsigned short* __restrict__ wt,
                                     int K, int N) {
  const int i = blockIdx.x * blockDim.x + threadIdx.x;
  if (i < N * K) {
    const int n  = i / K;
    const int kk = i % K;
    wt[i] = f2bf(w[(size_t)kk * N + n]);
  }
}

// ---------------------------------------------------------------------------
extern "C" void kernel_launch(void* const* d_in, const int* in_sizes, int n_in,
                              void* d_out, int out_size, void* d_ws, size_t ws_size,
                              hipStream_t stream) {
  (void)in_sizes; (void)n_in; (void)out_size; (void)ws_size;
  const float* x     = (const float*)d_in[0];
  const float* w_qkv = (const float*)d_in[1];
  const float* b_qkv = (const float*)d_in[2];
  const float* w_out = (const float*)d_in[3];
  const float* b_out = (const float*)d_in[4];
  float* out = (float*)d_out;

  // workspace layout (bf16 halves), total ~48 MB
  unsigned short* xb    = (unsigned short*)d_ws;               // [4096,1024]
  unsigned short* wqkvT = xb    + (size_t)MTOT * DIM;          // [3072,1024]
  unsigned short* woT   = wqkvT + (size_t)NQKV * DIM;          // [1024,1024]
  unsigned short* qkv   = woT   + (size_t)DIM * DIM;           // 3x[B,H,N,64]
  const size_t QS = (size_t)BATCH * NHEAD * SEQ * HD;          // 4194304
  unsigned short* ybuf  = qkv + 3 * QS;                        // [4096,1024]

  cvt_kernel<<<(MTOT * DIM) / 256, 256, 0, stream>>>(x, xb, MTOT * DIM);
  transpose_cvt_kernel<<<(NQKV * DIM) / 256, 256, 0, stream>>>(w_qkv, wqkvT, DIM, NQKV);
  transpose_cvt_kernel<<<(DIM * DIM) / 256, 256, 0, stream>>>(w_out, woT, DIM, DIM);

  // qkv = x @ w_qkv + b_qkv, scattered into q/k/v [B,H,N,64] bf16
  gemm_bf16_kernel<<<(MTOT / TM) * (NQKV / TN), 256, 0, stream>>>(
      xb, wqkvT, b_qkv, nullptr, qkv, MTOT, NQKV, DIM, 1);

  // flash attention -> ybuf [B,N,DIM] bf16
  attn_kernel<<<BATCH * NHEAD * (SEQ / 128), 256, 0, stream>>>(
      qkv, qkv + QS, qkv + 2 * QS, ybuf);

  // out = ybuf @ w_out + b_out (fp32)
  gemm_bf16_kernel<<<(MTOT / TM) * (DIM / TN), 256, 0, stream>>>(
      ybuf, woT, b_out, out, nullptr, MTOT, DIM, DIM, 0);
}
